// SableNetwork_40372692583281
// MI455X (gfx1250) — compile-verified, matrix operand order checked by reading, not run on hardware
//
#include <hip/hip_runtime.h>
#include <hip/hip_bf16.h>

// ---------------------------------------------------------------------------
// Sable encoder forward for MI455X (gfx1250), wave32 + WMMA f16 (f32 accum),
// with CDNA5 async global->LDS staging (ASYNCcnt) for pure-copy tiles.
//
// Assumed d_in flattening (setup_inputs() insertion order, recursive):
//  0 obs(8,1024,128) 1 hstate(8,8,2,32,32) 2 step_count(8,1024) int32
//  3 obs_ln(128) 4 W_obs(128,256)
//  per block i (base 5+10*i): ln1,ln2,Wq,Wk,Wv,Wg,Wo,W_gate,W_linear,W_output
//  head (base 25): W1(256,256), b1(256), ln(256), W2(256,1), b2(1)
// Output: value (8,1024,1) fp32 -> 8192 floats.
// ---------------------------------------------------------------------------

typedef __attribute__((ext_vector_type(16))) _Float16 v16h;
typedef __attribute__((ext_vector_type(8)))  float    v8f;

#define R_TOT 8192      // B*S
#define SEQ   1024
#define EMB   256
#define NHEAD 8
#define DH    32

union U32H2 { unsigned int u; _Float16 h[2]; };

__device__ __forceinline__ int kkof(int v, int hi) {
  // K index of first half held in half-register v for this lane-half (ISA 7.12.2)
  return ((v >> 2) << 4) + (hi << 3) + ((v & 3) << 1);
}

__device__ __forceinline__ float gelu_f(float x) {
  float x3 = x * x * x;
  return 0.5f * x * (1.0f + tanhf(0.7978845608028654f * (x + 0.044715f * x3)));
}
__device__ __forceinline__ float silu_f(float x) {
  return x / (1.0f + __expf(-x));
}

// ---- CDNA5 async global->LDS copies (ASYNCcnt path) -----------------------
// LDS aperture check only inspects addr[63:32]; addr[31:0] of a flat shared
// pointer is the wave-relative LDS byte offset, which is what VDST wants.
__device__ __forceinline__ void async_b128(void* lds, const void* g) {
  asm volatile("global_load_async_to_lds_b128 %0, %1, off"
               :: "v"((unsigned)(uintptr_t)lds), "v"(g) : "memory");
}
__device__ __forceinline__ void async_b64(void* lds, const void* g) {
  asm volatile("global_load_async_to_lds_b64 %0, %1, off"
               :: "v"((unsigned)(uintptr_t)lds), "v"(g) : "memory");
}
__device__ __forceinline__ void wait_async0() {
  asm volatile("s_wait_asynccnt 0" ::: "memory");
}

enum { EPI_NONE = 0, EPI_GELU = 1, EPI_SILU = 2 };

// ---------------------------------------------------------------------------
// Generic tiled GEMM: C[M,N] = epi(alpha * A16[M,K] @ W[K,N] + bias) (+resid)
// Block = 256 threads = 8 waves -> 64x32 C tile (waves 4x2, 16x16 each).
// A tile is staged with async global->LDS b128; W tile needs f32->f16 so it
// stays on the VALU path.
// ---------------------------------------------------------------------------
template<int EPI, bool OUT16, bool RESID>
__global__ __launch_bounds__(256)
void gemm16(const _Float16* __restrict__ A, const float* __restrict__ W,
            const float* __restrict__ bias, float alpha,
            float* __restrict__ Cf, _Float16* __restrict__ Ch,
            int M, int K, int N)
{
  __shared__ alignas(16) _Float16 At[64][32];   // row-major activation tile
  __shared__ alignas(16) _Float16 Bt[32][32];   // Bt[n][k] (transposed weights)

  const int tid  = threadIdx.x;
  const int lane = tid & 31, wave = tid >> 5;
  const int hi = lane >> 4, lm = lane & 15;
  const int wm = wave >> 1, wn = wave & 1;
  const int m0 = blockIdx.x * 64, n0 = blockIdx.y * 32;

  v8f acc = {};
  for (int k0 = 0; k0 < K; k0 += 32) {
    __syncthreads();
    { // stage A: 64 rows x 32 halves, async copy (16B per lane)
      int r = tid >> 2, c8 = (tid & 3) * 8;
      async_b128(&At[r][c8], A + (size_t)(m0 + r) * K + k0 + c8);
    }
    { // stage W -> f16, transposed: Bt[n][k]
      int kk = tid >> 3, n4 = (tid & 7) * 4;
      float4 wv = *(const float4*)(W + (size_t)(k0 + kk) * N + n0 + n4);
      Bt[n4 + 0][kk] = (_Float16)wv.x;
      Bt[n4 + 1][kk] = (_Float16)wv.y;
      Bt[n4 + 2][kk] = (_Float16)wv.z;
      Bt[n4 + 3][kk] = (_Float16)wv.w;
    }
    if (k0 + 32 < K) {  // global_prefetch_b8 for next W tile
      __builtin_prefetch(W + (size_t)(k0 + 32 + (tid >> 3)) * N + n0, 0, 1);
    }
    wait_async0();
    __syncthreads();

    v16h a, b;
#pragma unroll
    for (int v = 0; v < 8; ++v) {
      int kk = kkof(v, hi);
      U32H2 ua; ua.u = *(const unsigned int*)&At[wm * 16 + lm][kk];
      a[2 * v] = ua.h[0]; a[2 * v + 1] = ua.h[1];
      U32H2 ub; ub.u = *(const unsigned int*)&Bt[wn * 16 + lm][kk];
      b[2 * v] = ub.h[0]; b[2 * v + 1] = ub.h[1];
    }
    acc = __builtin_amdgcn_wmma_f32_16x16x32_f16(false, a, false, b,
                                                 (short)0, acc, false, false);
  }

  const int col = n0 + wn * 16 + lm;
  const float bv = bias ? bias[col] : 0.0f;
#pragma unroll
  for (int r = 0; r < 8; ++r) {
    int row = m0 + wm * 16 + r + 8 * hi;
    float v = acc[r] * alpha + bv;
    if (EPI == EPI_GELU) v = gelu_f(v);
    else if (EPI == EPI_SILU) v = silu_f(v);
    size_t idx = (size_t)row * N + col;
    if (RESID) v += Cf[idx];
    if (OUT16) Ch[idx] = (_Float16)v;
    else       Cf[idx] = v;
  }
}

// ---------------------------------------------------------------------------
// Row-wise RMS norm + cast to f16. One block per row, blockDim = K.
// ---------------------------------------------------------------------------
__global__ void rmsnorm16(const float* __restrict__ x, const float* __restrict__ w,
                          _Float16* __restrict__ o, int K)
{
  extern __shared__ float red[];
  const int row = blockIdx.x, t = threadIdx.x;
  float v = x[(size_t)row * K + t];
  red[t] = v * v;
  __syncthreads();
  for (int s = K >> 1; s > 0; s >>= 1) {
    if (t < s) red[t] += red[t + s];
    __syncthreads();
  }
  float scale = rsqrtf(red[0] / (float)K + 1e-6f);
  o[(size_t)row * K + t] = (_Float16)(v * scale * w[t]);
}

__global__ __launch_bounds__(256)
void cvt16(const float* __restrict__ a, _Float16* __restrict__ o, int n)
{
  int i = blockIdx.x * 256 + threadIdx.x;
  if (i < n) o[i] = (_Float16)a[i];
}

__global__ __launch_bounds__(256)
void mul16(const float* __restrict__ a, const float* __restrict__ b,
           _Float16* __restrict__ o, int n)
{
  int i = blockIdx.x * 256 + threadIdx.x;
  if (i < n) o[i] = (_Float16)(a[i] * b[i]);
}

// ---------------------------------------------------------------------------
// RoPE on q,k (per head, half=16) using step_count; emits f16 operands.
// ---------------------------------------------------------------------------
__global__ __launch_bounds__(256)
void rope_kernel(const float* __restrict__ q, const float* __restrict__ k,
                 const int* __restrict__ steps,
                 _Float16* __restrict__ q16o, _Float16* __restrict__ k16o)
{
  const int row = blockIdx.x, c = threadIdx.x;
  const int h = c >> 5, d = c & 31, i = d & 15;
  const float t = (float)steps[row];
  const float freq = exp2f(-(float)i * 0.8304820237218405f); // log2(1e4)/16
  float sn, cs;
  __sincosf(t * freq, &sn, &cs);
  const size_t base = (size_t)row * EMB + h * DH;
  float q1 = q[base + i], q2 = q[base + i + 16];
  float k1 = k[base + i], k2 = k[base + i + 16];
  float qo = (d < 16) ? (q1 * cs - q2 * sn) : (q1 * sn + q2 * cs);
  float ko = (d < 16) ? (k1 * cs - k2 * sn) : (k1 * sn + k2 * cs);
  q16o[(size_t)row * EMB + c] = (_Float16)qo;
  k16o[(size_t)row * EMB + c] = (_Float16)ko;
}

// ---------------------------------------------------------------------------
// Retention: out = (QK^T ⊙ κ^|n-m|) V + (Q ⊙ κ^(n+1)) H, then group-norm(Dh).
// Grid: (S/128, B*H). 8 waves, each owns 16 query rows x 32 (Dh) output.
// K tile staged via async global->LDS b64; V tile transposed on the fly.
// ---------------------------------------------------------------------------
__global__ __launch_bounds__(256)
void retention_kernel(const _Float16* __restrict__ q16, const _Float16* __restrict__ k16,
                      const _Float16* __restrict__ v16, const float* __restrict__ hstate,
                      float* __restrict__ gn, int blk)
{
  __shared__ alignas(16) _Float16 Kt [32][32];       // Kt[key][feat]
  __shared__ alignas(16) _Float16 VtT[32][32];       // VtT[feat][key]
  __shared__ alignas(16) _Float16 Sscr[8][16][32];   // per-wave scores (f16)
  __shared__ alignas(16) float    Oscr[8][16][32];   // per-wave out (f32)

  const int tid = threadIdx.x;
  const int lane = tid & 31, wave = tid >> 5;
  const int hi = lane >> 4, lm = lane & 15;
  const int b = blockIdx.y >> 3, h = blockIdx.y & 7;
  const int mrow = blockIdx.x * 128 + wave * 16;

  const float kappa = 1.0f - exp2f(-5.0f - (float)h);
  const float ch = log2f(kappa);

  // Q operand (A layout): lane holds row mrow+lm, 16 K-halves
  v16h qa;
  {
    const int m = mrow + lm;
    const _Float16* qp = q16 + ((size_t)(b * SEQ + m)) * EMB + h * DH;
#pragma unroll
    for (int v = 0; v < 8; ++v) {
      int kk = kkof(v, hi);
      U32H2 u; u.u = *(const unsigned int*)(qp + kk);
      qa[2 * v] = u.h[0]; qa[2 * v + 1] = u.h[1];
    }
  }

  v8f o0 = {}, o1 = {};

  for (int j = 0; j < SEQ / 32; ++j) {
    const int n0w = j * 32;
    __syncthreads();
    { // cooperative stage of 32 keys: K async copy, V transposed manually
      int r = tid >> 3, c4 = (tid & 7) * 4;
      const size_t gr = (size_t)(b * SEQ + n0w + r);
      async_b64(&Kt[r][c4], k16 + gr * EMB + h * DH + c4);
      uint2 vv = *(const uint2*)(v16 + gr * EMB + h * DH + c4);
      U32H2 u0, u1; u0.u = vv.x; u1.u = vv.y;
      VtT[c4 + 0][r] = u0.h[0]; VtT[c4 + 1][r] = u0.h[1];
      VtT[c4 + 2][r] = u1.h[0]; VtT[c4 + 3][r] = u1.h[1];
    }
    wait_async0();
    __syncthreads();

#pragma unroll
    for (int jj = 0; jj < 2; ++jj) {
      v16h bk;
#pragma unroll
      for (int v = 0; v < 8; ++v) {
        int kk = kkof(v, hi);
        U32H2 u; u.u = *(const unsigned int*)&Kt[jj * 16 + lm][kk];
        bk[2 * v] = u.h[0]; bk[2 * v + 1] = u.h[1];
      }
      v8f zc = {};
      v8f s = __builtin_amdgcn_wmma_f32_16x16x32_f16(false, qa, false, bk,
                                                     (short)0, zc, false, false);
      // decay κ^|m-n| and relayout C->A via per-wave LDS scratch
#pragma unroll
      for (int r = 0; r < 8; ++r) {
        int mg = mrow + r + 8 * hi;
        int kn = n0w + jj * 16 + lm;
        float d = exp2f(ch * fabsf((float)(mg - kn)));
        Sscr[wave][r + 8 * hi][jj * 16 + lm] = (_Float16)(s[r] * d);
      }
    }

    v16h a2, bv0, bv1;
#pragma unroll
    for (int v = 0; v < 8; ++v) {
      int kk = kkof(v, hi);
      U32H2 u; u.u = *(const unsigned int*)&Sscr[wave][lm][kk];
      a2[2 * v] = u.h[0]; a2[2 * v + 1] = u.h[1];
      U32H2 w0; w0.u = *(const unsigned int*)&VtT[lm][kk];
      bv0[2 * v] = w0.h[0]; bv0[2 * v + 1] = w0.h[1];
      U32H2 w1; w1.u = *(const unsigned int*)&VtT[16 + lm][kk];
      bv1[2 * v] = w1.h[0]; bv1[2 * v + 1] = w1.h[1];
    }
    o0 = __builtin_amdgcn_wmma_f32_16x16x32_f16(false, a2, false, bv0,
                                                (short)0, o0, false, false);
    o1 = __builtin_amdgcn_wmma_f32_16x16x32_f16(false, a2, false, bv1,
                                                (short)0, o1, false, false);
  }

  // cross term: (Q ⊙ κ^(m+1)) @ hstate[b,h,blk]
  {
    const int m = mrow + lm;
    const float qd = exp2f(ch * (float)(m + 1));
    v16h ad;
#pragma unroll
    for (int i = 0; i < 16; ++i) ad[i] = qa[i] * (_Float16)qd;

    const float* hs = hstate + (((size_t)(b * NHEAD + h)) * 2 + blk) * (DH * DH);
    v16h bh0, bh1;
#pragma unroll
    for (int v = 0; v < 8; ++v) {
      int kk = kkof(v, hi);
      bh0[2 * v]     = (_Float16)hs[(kk + 0) * DH + lm];
      bh0[2 * v + 1] = (_Float16)hs[(kk + 1) * DH + lm];
      bh1[2 * v]     = (_Float16)hs[(kk + 0) * DH + 16 + lm];
      bh1[2 * v + 1] = (_Float16)hs[(kk + 1) * DH + 16 + lm];
    }
    o0 = __builtin_amdgcn_wmma_f32_16x16x32_f16(false, ad, false, bh0,
                                                (short)0, o0, false, false);
    o1 = __builtin_amdgcn_wmma_f32_16x16x32_f16(false, ad, false, bh1,
                                                (short)0, o1, false, false);
  }

  // epilogue: group-norm over Dh per row, write gn (fp32)
#pragma unroll
  for (int r = 0; r < 8; ++r) {
    Oscr[wave][r + 8 * hi][lm]      = o0[r];
    Oscr[wave][r + 8 * hi][16 + lm] = o1[r];
  }
  if (lane < 16) {
    const int m = mrow + lane;
    float mu = 0.0f;
#pragma unroll
    for (int c = 0; c < DH; ++c) mu += Oscr[wave][lane][c];
    mu *= (1.0f / DH);
    float var = 0.0f;
#pragma unroll
    for (int c = 0; c < DH; ++c) {
      float d = Oscr[wave][lane][c] - mu;
      var += d * d;
    }
    var *= (1.0f / DH);
    float inv = rsqrtf(var + 1e-6f);
    float* gp = gn + ((size_t)(b * SEQ + m)) * EMB + h * DH;
#pragma unroll
    for (int c = 0; c < DH; ++c) gp[c] = (Oscr[wave][lane][c] - mu) * inv;
  }
}

// ---------------------------------------------------------------------------
// Head tail: per row, y' = rms_norm(y, ln); out = y' @ W2 + b2 (N=1).
// ---------------------------------------------------------------------------
__global__ __launch_bounds__(256)
void head_final(const float* __restrict__ y, const float* __restrict__ lnw,
                const float* __restrict__ W2, const float* __restrict__ b2,
                float* __restrict__ out)
{
  __shared__ float red[256];
  const int row = blockIdx.x, t = threadIdx.x;
  float v = y[(size_t)row * EMB + t];
  red[t] = v * v;
  __syncthreads();
  for (int s = 128; s > 0; s >>= 1) { if (t < s) red[t] += red[t + s]; __syncthreads(); }
  float scale = rsqrtf(red[0] / (float)EMB + 1e-6f);
  __syncthreads();
  red[t] = v * scale * lnw[t] * W2[t];
  __syncthreads();
  for (int s = 128; s > 0; s >>= 1) { if (t < s) red[t] += red[t + s]; __syncthreads(); }
  if (t == 0) out[row] = red[0] + b2[0];
}

// ---------------------------------------------------------------------------
extern "C" void kernel_launch(void* const* d_in, const int* in_sizes, int n_in,
                              void* d_out, int out_size, void* d_ws, size_t ws_size,
                              hipStream_t stream)
{
  (void)in_sizes; (void)n_in; (void)out_size; (void)ws_size;

  const float* obs    = (const float*)d_in[0];
  const float* hstate = (const float*)d_in[1];
  const int*   steps  = (const int*)  d_in[2];
  const float* obs_ln = (const float*)d_in[3];
  const float* W_obs  = (const float*)d_in[4];
  const float* hW1 = (const float*)d_in[25];
  const float* hb1 = (const float*)d_in[26];
  const float* hln = (const float*)d_in[27];
  const float* hW2 = (const float*)d_in[28];
  const float* hb2 = (const float*)d_in[29];

  char* ws = (char*)d_ws;
  float*    X   = (float*)   (ws + (size_t) 0u);         // 8 MB
  float*    P0  = (float*)   (ws + (size_t) 8u * (1u<<20));
  float*    P1  = (float*)   (ws + (size_t)16u * (1u<<20));
  float*    P2  = (float*)   (ws + (size_t)24u * (1u<<20));
  _Float16* A16 = (_Float16*)(ws + (size_t)32u * (1u<<20));
  _Float16* H0  = (_Float16*)(ws + (size_t)36u * (1u<<20)); // v16
  _Float16* H1  = (_Float16*)(ws + (size_t)40u * (1u<<20)); // q16 / y16
  _Float16* H2  = (_Float16*)(ws + (size_t)44u * (1u<<20)); // k16

  const dim3 gGemm(R_TOT / 64, EMB / 32);
  const dim3 gRet(SEQ / 128, 8 * NHEAD);
  const int  nElem = R_TOT * EMB;
  const float kscale = 0.17677669529663687f; // 1/sqrt(32)

  // obs_rep = gelu(rms_norm(obs) @ W_obs)
  rmsnorm16<<<R_TOT, 128, 128 * 4, stream>>>(obs, obs_ln, A16, 128);
  gemm16<EPI_GELU, false, false><<<gGemm, 256, 0, stream>>>(
      A16, W_obs, nullptr, 1.0f, X, nullptr, R_TOT, 128, EMB);

  for (int blk = 0; blk < 2; ++blk) {
    const int base = 5 + blk * 10;
    const float* ln1  = (const float*)d_in[base + 0];
    const float* ln2  = (const float*)d_in[base + 1];
    const float* Wq   = (const float*)d_in[base + 2];
    const float* Wk   = (const float*)d_in[base + 3];
    const float* Wv   = (const float*)d_in[base + 4];
    const float* Wg   = (const float*)d_in[base + 5];
    const float* Wo   = (const float*)d_in[base + 6];
    const float* Wga  = (const float*)d_in[base + 7];
    const float* Wli  = (const float*)d_in[base + 8];
    const float* Wou  = (const float*)d_in[base + 9];

    rmsnorm16<<<R_TOT, 256, 256 * 4, stream>>>(X, ln1, A16, EMB);
    gemm16<EPI_NONE, false, false><<<gGemm, 256, 0, stream>>>(
        A16, Wq, nullptr, 1.0f, P0, nullptr, R_TOT, EMB, EMB);        // q32
    gemm16<EPI_NONE, false, false><<<gGemm, 256, 0, stream>>>(
        A16, Wk, nullptr, kscale, P1, nullptr, R_TOT, EMB, EMB);      // k32
    gemm16<EPI_NONE, true,  false><<<gGemm, 256, 0, stream>>>(
        A16, Wv, nullptr, 1.0f, nullptr, H0, R_TOT, EMB, EMB);        // v16
    gemm16<EPI_SILU, false, false><<<gGemm, 256, 0, stream>>>(
        A16, Wg, nullptr, 1.0f, P2, nullptr, R_TOT, EMB, EMB);        // gate

    rope_kernel<<<R_TOT, 256, 0, stream>>>(P0, P1, steps, H1, H2);    // q16,k16
    retention_kernel<<<gRet, 256, 0, stream>>>(H1, H2, H0, hstate, P0, blk); // gn
    mul16<<<nElem / 256, 256, 0, stream>>>(P2, P0, H1, nElem);        // gate*gn
    gemm16<EPI_NONE, false, true><<<gGemm, 256, 0, stream>>>(
        H1, Wo, nullptr, 1.0f, X, nullptr, R_TOT, EMB, EMB);          // x += .@Wo

    rmsnorm16<<<R_TOT, 256, 256 * 4, stream>>>(X, ln2, A16, EMB);
    gemm16<EPI_SILU, false, false><<<gGemm, 256, 0, stream>>>(
        A16, Wga, nullptr, 1.0f, P0, nullptr, R_TOT, EMB, EMB);
    gemm16<EPI_NONE, false, false><<<gGemm, 256, 0, stream>>>(
        A16, Wli, nullptr, 1.0f, P1, nullptr, R_TOT, EMB, EMB);
    mul16<<<nElem / 256, 256, 0, stream>>>(P0, P1, H1, nElem);
    gemm16<EPI_NONE, false, true><<<gGemm, 256, 0, stream>>>(
        H1, Wou, nullptr, 1.0f, X, nullptr, R_TOT, EMB, EMB);         // x += ffn
  }

  // head: y = gelu(x @ W1 + b1); out = rms_norm(y, ln) @ W2 + b2
  cvt16<<<nElem / 256, 256, 0, stream>>>(X, A16, nElem);
  gemm16<EPI_GELU, false, false><<<gGemm, 256, 0, stream>>>(
      A16, hW1, hb1, 1.0f, P0, nullptr, R_TOT, EMB, EMB);
  head_final<<<R_TOT, 256, 0, stream>>>(P0, hln, hW2, hb2, (float*)d_out);
}
